// DistributedRealVectorSHT_63848983822763
// MI455X (gfx1250) — matrix operand order repaired and checked by reading, not compiled, not run on hardware
//
#include <hip/hip_runtime.h>
#include <math.h>

// ---------------------------------------------------------------------------
// DistributedRealVectorSHT for MI455X (gfx1250, wave32, WMMA f16 16x16x32)
//
// Stage 0: convert x (f32) -> f16 (K-pad 720->736); generate DFT matrix f16;
//          zero Y plane buffer (provides lat-pad zeros).
// Stage 1: DFT as GEMM  Y[row,col] = sum_n x[row,n] * D[col,n]  (WMMA f16).
//          Each wave computes an nt-PAIR: one A fragment feeds two WMMAs so
//          each s_wait_loadcnt covers two matrix ops (2 accumulator chains).
//          Epilogue scatters into plane layout Y[m][q][b][lat],
//          q in {y0r, y0i, y1r, -y1i}.
// Stage 2: per-m GEMMs (WMMA f16):
//            out0r = q0*w0 + q3*w1        out0i = q1*w0 + q2*w1
//            out1r = -(q1*w1 + q2*w0)     out1i = q0*w1 + q3*w0
//          A planes staged to LDS via global_load_async_to_lds_b128
//          (ASYNCcnt); weight tiles double-buffered: next tile's f32 global
//          loads are issued before the WMMA loop (LOADcnt) while compute
//          waits only on DScnt, then converted to f16 into the spare buffer.
//
// Workspace: Axh 33.9MB + Bd 1.08MB + Yw 35.5MB = ~70.5 MB.
// ---------------------------------------------------------------------------

typedef _Float16 v16h __attribute__((ext_vector_type(16)));
typedef _Float16 v8h  __attribute__((ext_vector_type(8)));
typedef float    v8f  __attribute__((ext_vector_type(8)));

#define NLAT 360
#define NLON 720
#define LMAX 360
#define MMAX 361
#define NB   32
#define ROWS (NB * 2 * NLAT)   // 23040 fft rows
#define K1   736               // padded NLON (23 k-tiles of 32)
#define NCOL 722               // 361 cos + 361 sin columns
#define NT1  46                // ceil(NCOL/16)
#define K2   384               // padded NLAT (12 k-tiles of 32)
#define NT2  23                // ceil(LMAX/16)
#define TWO_PI 6.28318530717958647692f

static constexpr size_t AXH_ELEMS = (size_t)ROWS * K1;          // 16,957,440 halves
static constexpr size_t BD_ELEMS  = (size_t)K1 * K1;            //    541,696 halves
static constexpr size_t YW_ELEMS  = (size_t)MMAX * 4 * NB * K2; // 17,743,872 halves

#define A_LDS_HALVES (4 * NB * K2)        // 49152 halves = 98304 B
#define W_LDS_HALVES (2 * 2 * 16 * K2)    // 24576 halves = 49152 B
#define SMEM_BYTES   ((A_LDS_HALVES + W_LDS_HALVES) * 2)  // 147456 B

// --- WMMA wrapper: D = A(16x32 f16) * B(32x16 f16) + C(16x16 f32) ----------
__device__ __forceinline__ v8f wmma_f16(v16h a, v16h b, v8f c) {
    return __builtin_amdgcn_wmma_f32_16x16x32_f16(
        /*neg_a=*/false, a, /*neg_b=*/false, b,
        /*c_mod=*/(short)0, c, /*reuse_a=*/false, /*reuse_b=*/false);
}

// A fragment from row-major [M][K] tile: lane L -> row L%16,
// halves (L/16)*8..+7 in VGPR0-3 and +16.. in VGPR4-7. Two B128 loads.
__device__ __forceinline__ v16h load_frag_a(const _Float16* __restrict__ p) {
    union { v16h v; v8h q[2]; } u;
    u.q[0] = *(const v8h*)(p);
    u.q[1] = *(const v8h*)(p + 16);
    return u.v;
}
// B fragment from row-major [N][K] tile (B^T): lane L -> column L%16,
// halves (L/16)*8..+7 in VGPR0-3 and +16.. in VGPR4-7. Two B128 loads.
__device__ __forceinline__ v16h load_frag_b(const _Float16* __restrict__ p) {
    union { v16h v; v8h q[2]; } u;
    u.q[0] = *(const v8h*)(p);
    u.q[1] = *(const v8h*)(p + 16);
    return u.v;
}

// --- CDNA5 async global->LDS copy (ASYNCcnt) --------------------------------
__device__ __forceinline__ void async_g2l_b128(unsigned lds_addr, unsigned gofs,
                                               const void* sbase) {
    asm volatile("global_load_async_to_lds_b128 %0, %1, %2"
                 :: "v"(lds_addr), "v"(gofs), "s"(sbase)
                 : "memory");
}
__device__ __forceinline__ void wait_async0() {
#if __has_builtin(__builtin_amdgcn_s_wait_asynccnt)
    __builtin_amdgcn_s_wait_asynccnt(0);
#else
    asm volatile("s_wait_asynccnt 0x0" ::: "memory");
#endif
}

// --- Stage 0a: x f32 -> f16, pad columns 720..735 with zeros ----------------
__global__ __launch_bounds__(256) void k_convert_x(const float* __restrict__ X,
                                                   _Float16* __restrict__ Axh) {
    size_t i = (size_t)blockIdx.x * 256 + threadIdx.x;    // vec-of-8 index
    if (i >= AXH_ELEMS / 8) return;
    size_t f = i * 8;
    int row = (int)(f / K1);
    int n0  = (int)(f % K1);
    v8h v;
    if (n0 < NLON) {                       // 720 % 8 == 0: no mixed vectors
        const float* p = X + (size_t)row * NLON + n0;
#pragma unroll
        for (int j = 0; j < 8; ++j) v[j] = (_Float16)p[j];
    } else {
#pragma unroll
        for (int j = 0; j < 8; ++j) v[j] = (_Float16)0.0f;
    }
    *(v8h*)(Axh + f) = v;
}

// --- Stage 0b: DFT matrix, row-major [col][n]: cos / sin * (2pi/720) --------
__global__ __launch_bounds__(256) void k_gen_dft(_Float16* __restrict__ Bd) {
    int i = blockIdx.x * 256 + threadIdx.x;
    if (i >= (int)BD_ELEMS) return;
    int col = i / K1;
    int n   = i % K1;
    float val = 0.0f;
    if (col < NCOL && n < NLON) {
        int mm = (col < MMAX) ? col : col - MMAX;
        int rr = (mm * n) % NLON;                 // exact phase in integers
        float th = (float)rr * (TWO_PI / NLON);
        float sc = TWO_PI / NLON;
        val = ((col < MMAX) ? __cosf(th) : __sinf(th)) * sc;
    }
    Bd[i] = (_Float16)val;
}

// --- Stage 0c: zero Y plane buffer ------------------------------------------
__global__ __launch_bounds__(256) void k_zero_yw(_Float16* __restrict__ Yw) {
    size_t i = (size_t)blockIdx.x * 256 + threadIdx.x;
    if (i >= YW_ELEMS / 8) return;
    v8h z = {};
    *(v8h*)(Yw + i * 8) = z;
}

// --- Stage 1: DFT GEMM, 23040 x 722 (K=720), 1 block per 16-row M-tile ------
// Each wave owns an nt-pair: one A fragment -> two WMMAs per k-tile.
__global__ __launch_bounds__(256) void k_dft_gemm(const _Float16* __restrict__ Axh,
                                                  const _Float16* __restrict__ Bd,
                                                  _Float16* __restrict__ Yw) {
    const int mt   = blockIdx.x;        // 0..1439
    const int lane = threadIdx.x & 31;
    const int wave = threadIdx.x >> 5;  // 8 waves
    const int r = lane & 15, h = lane >> 4;

    const _Float16* aBase = Axh + (size_t)(mt * 16 + r) * K1 + h * 8;

    auto epilogue = [&](int nt, const v8f& acc) {
        int col = nt * 16 + r;           // D[M = d + h*8, N = r]
        if (col < NCOL) {
            bool isSin = (col >= MMAX);
            int mm = isSin ? col - MMAX : col;
#pragma unroll
            for (int d = 0; d < 8; ++d) {
                int row = mt * 16 + h * 8 + d;     // (b, c, lat)
                int b   = row / (2 * NLAT);
                int rem = row - b * (2 * NLAT);
                int c   = rem / NLAT;
                int lat = rem - c * NLAT;
                float v = acc[d];
                int p; float s;
                if (c == 0) { p = isSin ? 1 : 0; s = isSin ? -v : v; } // y0r, y0i=-S
                else        { p = isSin ? 3 : 2; s = v;             } // y1r, -y1i=+S
                Yw[((size_t)(mm * 4 + p) * NB + b) * K2 + lat] = (_Float16)s;
            }
        }
    };

    for (int pr = wave; pr < NT1 / 2; pr += 8) {   // 23 nt-pairs over 8 waves
        const int nt0 = 2 * pr, nt1 = 2 * pr + 1;
        const _Float16* b0 = Bd + (size_t)(nt0 * 16 + r) * K1 + h * 8;
        const _Float16* b1 = Bd + (size_t)(nt1 * 16 + r) * K1 + h * 8;
        v8f acc0 = {}, acc1 = {};
#pragma unroll
        for (int kt = 0; kt < 23; ++kt) {
            v16h af  = load_frag_a(aBase + kt * 32);
            v16h bf0 = load_frag_b(b0 + kt * 32);
            v16h bf1 = load_frag_b(b1 + kt * 32);
            acc0 = wmma_f16(af, bf0, acc0);
            acc1 = wmma_f16(af, bf1, acc1);
        }
        epilogue(nt0, acc0);
        epilogue(nt1, acc1);
    }
}

// --- Stage 2: per-(m, nt-half) einsum GEMMs ---------------------------------
__global__ __launch_bounds__(256) void k_sht_gemm(const _Float16* __restrict__ Yw,
                                                  const float* __restrict__ Wsrc,
                                                  float* __restrict__ Out) {
    extern __shared__ char smem[];
    _Float16* Ash = (_Float16*)smem;                           // [4][32][384]
    _Float16* Wt  = (_Float16*)(smem + A_LDS_HALVES * 2);      // [2][2][16][384]

    const int m   = blockIdx.x;          // 0..360
    const int nt0 = blockIdx.y * 12;     // nt range [nt0, ntEnd)
    const int ntEnd = (blockIdx.y == 0) ? 12 : NT2;
    const int tid  = threadIdx.x;
    const int lane = tid & 31;
    const int wave = tid >> 5;
    const int o     = wave & 3;          // 0=out0r 1=out0i 2=out1r 3=out1i
    const int mtile = wave >> 2;         // 0 or 1
    const int r = lane & 15, h = lane >> 4;

    // product table: out[o] = q[QA]*w[WA] + q[QB]*w[WB]  (out1r negated at store)
    const int QA[4] = {0, 1, 1, 0};
    const int WA[4] = {0, 0, 1, 1};
    const int QB[4] = {3, 2, 2, 3};
    const int WB[4] = {1, 1, 0, 0};
    const int qa = QA[o], wa = WA[o], qb = QB[o], wb = WB[o];

    // ---- Async-stage the four Y planes for this m into LDS (ASYNCcnt) ----
    {
        const void* gsrc = (const void*)(Yw + (size_t)m * 4 * NB * K2);
        unsigned ldsBase = (unsigned)(uintptr_t)Ash;
        for (int f = tid; f < (A_LDS_HALVES * 2) / 16; f += 256)
            async_g2l_b128(ldsBase + f * 16, (unsigned)(f * 16), gsrc);
        wait_async0();
        __syncthreads();
    }

    // Weight-tile prefetch helpers (6 vec8 per thread per tile).
    float pf[6][8];
    auto prefetchW = [&](int nt) {
#pragma unroll
        for (int j = 0; j < 6; ++j) {
            int f   = (j * 256 + tid) * 8;
            int c   = f / (16 * K2);
            int rem = f - c * (16 * K2);
            int l   = rem / K2;
            int k0  = rem - l * K2;
            int gl  = nt * 16 + l;
            if (gl < LMAX && k0 < NLAT) {
                const float* wp = Wsrc + (((size_t)c * MMAX + m) * LMAX + gl) * NLAT + k0;
#pragma unroll
                for (int jj = 0; jj < 8; ++jj) pf[j][jj] = wp[jj];
            } else {
#pragma unroll
                for (int jj = 0; jj < 8; ++jj) pf[j][jj] = 0.0f;
            }
        }
    };
    auto commitW = [&](int buf) {
#pragma unroll
        for (int j = 0; j < 6; ++j) {
            int f   = (j * 256 + tid) * 8;
            int c   = f / (16 * K2);
            int rem = f - c * (16 * K2);
            int l   = rem / K2;
            int k0  = rem - l * K2;
            v8h v;
#pragma unroll
            for (int jj = 0; jj < 8; ++jj) v[jj] = (_Float16)pf[j][jj];
            *(v8h*)(Wt + ((size_t)(buf * 2 + c) * 16 + l) * K2 + k0) = v;
        }
    };

    const _Float16* aA = Ash + ((size_t)qa * NB + mtile * 16 + r) * K2 + h * 8;
    const _Float16* aB = Ash + ((size_t)qb * NB + mtile * 16 + r) * K2 + h * 8;

    // Prime first weight tile.
    prefetchW(nt0);
    commitW(nt0 & 1);
    __syncthreads();

    for (int nt = nt0; nt < ntEnd; ++nt) {
        const int buf = nt & 1;
        // Issue next tile's f32 global loads now (LOADcnt); the WMMA loop
        // below touches only LDS (DScnt), so these stream during compute.
        if (nt + 1 < ntEnd) prefetchW(nt + 1);

        const _Float16* bA = Wt + ((size_t)(buf * 2 + wa) * 16 + r) * K2 + h * 8;
        const _Float16* bB = Wt + ((size_t)(buf * 2 + wb) * 16 + r) * K2 + h * 8;
        v8f accA = {}, accB = {};
#pragma unroll
        for (int kt = 0; kt < 12; ++kt) {
            v16h af0 = load_frag_a(aA + kt * 32);
            v16h bf0 = load_frag_b(bA + kt * 32);
            accA = wmma_f16(af0, bf0, accA);
            v16h af1 = load_frag_a(aB + kt * 32);
            v16h bf1 = load_frag_b(bB + kt * 32);
            accB = wmma_f16(af1, bf1, accB);
        }

        // Store: out layout (b, oc, l, m) complex -> float pairs.
        int l = nt * 16 + r;
        if (l < LMAX) {
            int oc = o >> 1, ri = o & 1;
            float sgn = (o == 2) ? -1.0f : 1.0f;
#pragma unroll
            for (int d = 0; d < 8; ++d) {
                int b = mtile * 16 + h * 8 + d;
                Out[((((size_t)b * 2 + oc) * LMAX + l) * MMAX + m) * 2 + ri] =
                    sgn * (accA[d] + accB[d]);
            }
        }

        // Convert + store next tile into the spare buffer (waits LOADcnt here).
        if (nt + 1 < ntEnd) commitW(buf ^ 1);
        __syncthreads();
    }
}

// ---------------------------------------------------------------------------
extern "C" void kernel_launch(void* const* d_in, const int* in_sizes, int n_in,
                              void* d_out, int out_size, void* d_ws, size_t ws_size,
                              hipStream_t stream) {
    const float* X = (const float*)d_in[0];   // (1,32,2,360,720) f32
    const float* W = (const float*)d_in[1];   // (2,361,360,360) f32
    float* Out = (float*)d_out;               // (1,32,2,360,361) complex64 as f32 pairs

    _Float16* Axh = (_Float16*)d_ws;
    _Float16* Bd  = Axh + AXH_ELEMS;
    _Float16* Yw  = Bd + BD_ELEMS;

    dim3 blk(256);
    {
        int g = (int)((AXH_ELEMS / 8 + 255) / 256);
        k_convert_x<<<g, blk, 0, stream>>>(X, Axh);
    }
    {
        int g = (int)((BD_ELEMS + 255) / 256);
        k_gen_dft<<<g, blk, 0, stream>>>(Bd);
    }
    {
        int g = (int)((YW_ELEMS / 8 + 255) / 256);
        k_zero_yw<<<g, blk, 0, stream>>>(Yw);
    }
    k_dft_gemm<<<ROWS / 16, blk, 0, stream>>>(Axh, Bd, Yw);
    k_sht_gemm<<<dim3(MMAX, 2), blk, SMEM_BYTES, stream>>>(Yw, W, Out);
}
